// CondensedEmbracementLayer_39152921870981
// MI455X (gfx1250) — matrix operand order repaired and checked by reading, not compiled, not run on hardware
//
#include <hip/hip_runtime.h>

#define BS  256
#define SEQ 512
#define H   768
#define K2  (2*H)
#define GK  768   // GEMM K dimension (both GEMMs)

typedef __attribute__((ext_vector_type(16))) __bf16 v16bf;
typedef __attribute__((ext_vector_type(8)))  float  v8f;

union FragU {
    v16bf v;
    uint4 q[2];
    unsigned short s[16];
};

__device__ __forceinline__ unsigned short f32_to_bf16_rne(float f) {
    unsigned u = __float_as_uint(f);
    u += 0x7FFFu + ((u >> 16) & 1u);         // round-to-nearest-even
    return (unsigned short)(u >> 16);
}

__device__ __forceinline__ unsigned rotl32(unsigned x, int r) {
    return (x << r) | (x >> (32 - r));
}

// Flat LDS addresses carry the wave-relative LDS offset in addr[31:0] (ISA 10.2).
__device__ __forceinline__ unsigned lds_off(const void* p) {
    return (unsigned)(unsigned long long)p;
}

// Async copy 32 bytes (2 x b128) global -> LDS; INST_OFFSET applies to both sides.
__device__ __forceinline__ void async_tile_pair(unsigned lds, const unsigned short* g) {
    asm volatile(
        "global_load_async_to_lds_b128 %0, %1, off\n\t"
        "global_load_async_to_lds_b128 %0, %1, off offset:16"
        :: "v"(lds), "v"(g) : "memory");
}
__device__ __forceinline__ void wait_async0() {
    asm volatile("s_wait_asynccnt 0" ::: "memory");
}

// JAX threefry2x32 with key = jax.random.key(42) -> (0, 42)
__device__ unsigned threefry2x32_42(unsigned c0, unsigned c1, int which) {
    const unsigned ks0 = 0u, ks1 = 42u, ks2 = 0x1BD11BDAu ^ 42u;
    unsigned x0 = c0 + ks0, x1 = c1 + ks1;
#define TF_R(r) { x0 += x1; x1 = rotl32(x1, (r)); x1 ^= x0; }
    TF_R(13) TF_R(15) TF_R(26) TF_R(6)   x0 += ks1; x1 += ks2 + 1u;
    TF_R(17) TF_R(29) TF_R(16) TF_R(24)  x0 += ks2; x1 += ks0 + 2u;
    TF_R(13) TF_R(15) TF_R(26) TF_R(6)   x0 += ks0; x1 += ks1 + 3u;
    TF_R(17) TF_R(29) TF_R(16) TF_R(24)  x0 += ks1; x1 += ks2 + 4u;
    TF_R(13) TF_R(15) TF_R(26) TF_R(6)   x0 += ks2; x1 += ks0 + 5u;
#undef TF_R
    return which ? x1 : x0;
}

// -------- Kernel 0: fused convert + transpose: W f32 [K2][H] -> Wt bf16 [H][K2]
__global__ void transpose_convert_w(const float* __restrict__ W,
                                    unsigned short* __restrict__ Wt) {
    __shared__ unsigned short tile[32][33];
    const int n0 = blockIdx.x * 32;      // H (column of W) dimension
    const int k0 = blockIdx.y * 32;      // K2 (row of W) dimension
    const int tx = threadIdx.x, ty = threadIdx.y;   // block (32, 8)
    for (int i = ty; i < 32; i += 8)
        tile[i][tx] = f32_to_bf16_rne(W[(k0 + i) * H + n0 + tx]);
    __syncthreads();
    for (int i = ty; i < 32; i += 8)
        Wt[(n0 + i) * K2 + k0 + tx] = tile[tx][i];
}

// -------- Kernel 1: lengths + threefry sample + gather + bf16 prep ------------
__global__ void prep_kernel(const float* __restrict__ tokens,
                            const float* __restrict__ cls,
                            const int* __restrict__ mask,
                            float* __restrict__ emb,
                            unsigned short* __restrict__ clsbf,
                            unsigned short* __restrict__ Sbf) {
    __shared__ int red[256];
    __shared__ int s_len;
    int b = blockIdx.x, tid = threadIdx.x;

    // leading-ones length = position of first zero (or SEQ)
    int fz = SEQ;
    for (int t = tid; t < SEQ; t += 256)
        if (mask[b * SEQ + t] == 0) fz = min(fz, t);
    red[tid] = fz; __syncthreads();
    for (int s = 128; s > 0; s >>= 1) {
        if (tid < s) red[tid] = min(red[tid], red[tid + s]);
        __syncthreads();
    }
    if (tid == 0) s_len = red[0];
    __syncthreads();
    int len = s_len;
    float flen = (float)len;

    const int half = (BS * H) / 2;   // 98304; n = 196608 is even -> two halves
    for (int j = tid; j < H; j += 256) {
        int f = b * H + j;
        unsigned c0, c1; int which;
        if (f < half) { c0 = (unsigned)f;          c1 = (unsigned)(f + half); which = 0; }
        else          { c0 = (unsigned)(f - half); c1 = (unsigned)f;          which = 1; }
        unsigned bits = threefry2x32_42(c0, c1, which);
        float u = __uint_as_float((bits >> 9) | 0x3F800000u) - 1.0f;   // [0,1)
        int idx = (int)(u * flen);
        int maxi = len - 1; if (maxi < 0) maxi = 0;
        if (idx > maxi) idx = maxi;

        float e = tokens[(size_t)b * SEQ * H + (size_t)idx * H + j];
        float c = cls[b * H + j];
        emb[b * H + j] = e;
        unsigned short cb = f32_to_bf16_rne(c);
        clsbf[b * H + j] = cb;
        Sbf[(2 * b) * H + j]     = cb;
        Sbf[(2 * b + 1) * H + j] = f32_to_bf16_rne(e);
    }
}

// -------- Kernels 2/3: bf16 WMMA GEMM, C[M,N] = A[M,GK] @ Bt[N,GK]^T ----------
// A row-major [M][lda], Bt row-major [N][ldb] (pre-transposed weights).
// 128x128 block tile, 8 waves (2x4); wave = 64x32 = 4x2 tiles of 16x16.
// LDS tiles double-buffered; filled by async global->LDS b128 copies.
// Two K-steps per loop iteration -> buffer indices are compile-time constants
// and prefetch is unconditional (final iteration overfetches one dead tile,
// covered by workspace padding).
__global__ __launch_bounds__(256, 1)
void gemm_bf16_kernel(const unsigned short* __restrict__ A, int lda,
                      const unsigned short* __restrict__ Bt, int ldb,
                      float* __restrict__ C, int N) {
    __shared__ unsigned short As[2][128 * 32];   // [m][k], 8 KB each
    __shared__ unsigned short Bs[2][128 * 32];   // [n][k], 8 KB each

    const int tid  = threadIdx.x;
    const int lane = tid & 31;
    const int w    = tid >> 5;
    const int wrow = (w >> 2) * 64;   // 0 / 64
    const int wcol = (w & 3) * 32;    // 0 / 32 / 64 / 96
    const int bm   = blockIdx.y * 128;
    const int bn   = blockIdx.x * 128;

    // tile loaders: 128 rows x 2 x 32-byte segments, 32 B per thread per tile
    const int lrow = tid >> 1, lseg = tid & 1;
    const unsigned short* ga = A  + (bm + lrow) * lda + lseg * 16;
    const unsigned short* gb = Bt + (bn + lrow) * ldb + lseg * 16;
    const unsigned la0 = lds_off(&As[0][lrow * 32 + lseg * 16]);
    const unsigned la1 = lds_off(&As[1][lrow * 32 + lseg * 16]);
    const unsigned lb0 = lds_off(&Bs[0][lrow * 32 + lseg * 16]);
    const unsigned lb1 = lds_off(&Bs[1][lrow * 32 + lseg * 16]);

    v8f acc[4][2] = {};

    const int lm  = lane & 15;
    const int kbA = (lane >> 4) * 16;  // A frag: lanes<16 K0..15, lanes>=16 K16..31
    const int kbB = (lane >> 4) * 16;  // B frag: lanes<16 K0..15, lanes>=16 K16..31

    const unsigned short* aF = &As[0][(wrow + lm) * 32 + kbA];
    const unsigned short* bF = &Bs[0][(wcol + lm) * 32 + kbB];
    const int bufStrideA = (int)(&As[1][0] - &As[0][0]);
    const int bufStrideB = (int)(&Bs[1][0] - &Bs[0][0]);

    auto compute = [&](int bufA, int bufB) {
        FragU bfrag[2];
#pragma unroll
        for (int nt = 0; nt < 2; nt++) {
            const unsigned short* p = bF + bufB * bufStrideB + nt * (16 * 32);
            bfrag[nt].q[0] = *(const uint4*)p;
            bfrag[nt].q[1] = *(const uint4*)(p + 8);
        }
#pragma unroll
        for (int mt = 0; mt < 4; mt++) {
            FragU af;
            const unsigned short* p = aF + bufA * bufStrideA + mt * (16 * 32);
            af.q[0] = *(const uint4*)p;
            af.q[1] = *(const uint4*)(p + 8);
#pragma unroll
            for (int nt = 0; nt < 2; nt++) {
                acc[mt][nt] = __builtin_amdgcn_wmma_f32_16x16x32_bf16(
                    false, af.v, false, bfrag[nt].v, (short)0, acc[mt][nt], false, false);
            }
        }
    };

    // prologue: fill buffer 0 with K-step 0
    async_tile_pair(la0, ga);
    async_tile_pair(lb0, gb);
    wait_async0();
    __syncthreads();

#pragma unroll 1
    for (int kt = 0; kt < GK; kt += 64) {
        // prefetch K-step kt+32 into buffer 1
        async_tile_pair(la1, ga + kt + 32);
        async_tile_pair(lb1, gb + kt + 32);
        compute(0, 0);
        wait_async0();
        __syncthreads();
        // prefetch K-step kt+64 into buffer 0 (overfetch on last iter: harmless)
        async_tile_pair(la0, ga + kt + 64);
        async_tile_pair(lb0, gb + kt + 64);
        compute(1, 1);
        wait_async0();
        __syncthreads();
    }

    // C/D layout: VGPR e -> row e (lanes 0-15) / row 8+e (lanes 16-31), col = lane%16
    const int rbase = (lane < 16) ? 0 : 8;
#pragma unroll
    for (int mt = 0; mt < 4; mt++) {
#pragma unroll
        for (int nt = 0; nt < 2; nt++) {
            const int row0 = bm + wrow + mt * 16 + rbase;
            const int col  = bn + wcol + nt * 16 + lm;
            float* cp = C + row0 * N + col;   // 32-bit index math
#pragma unroll
            for (int e = 0; e < 8; e++)
                cp[e * N] = acc[mt][nt][e];
        }
    }
}

// -------- Kernel 4: energy + softmax(2) + blend -------------------------------
__global__ void finalize_kernel(const float* __restrict__ P,
                                const float* __restrict__ Q,
                                const float* __restrict__ bias,
                                const float* __restrict__ v,
                                const float* __restrict__ cls,
                                const float* __restrict__ emb,
                                float* __restrict__ out) {
    __shared__ float r0[256], r1[256];
    __shared__ float sa0, sa1;
    int b = blockIdx.x, tid = threadIdx.x;
    float s0 = 0.f, s1 = 0.f;
    for (int k = tid; k < H; k += 256) {
        float base = P[b * H + k] + bias[k];
        float vk = v[k];
        s0 += vk * tanhf(base + Q[(2 * b) * H + k]);
        s1 += vk * tanhf(base + Q[(2 * b + 1) * H + k]);
    }
    r0[tid] = s0; r1[tid] = s1; __syncthreads();
    for (int s = 128; s > 0; s >>= 1) {
        if (tid < s) { r0[tid] += r0[tid + s]; r1[tid] += r1[tid + s]; }
        __syncthreads();
    }
    if (tid == 0) {
        float t0 = r0[0], t1 = r1[0];
        float m = fmaxf(t0, t1);
        float e0 = __expf(t0 - m), e1 = __expf(t1 - m);
        float inv = 1.0f / (e0 + e1);
        sa0 = e0 * inv; sa1 = e1 * inv;
    }
    __syncthreads();
    float a0 = sa0, a1 = sa1;
    for (int k = tid; k < H; k += 256)
        out[b * H + k] = a0 * cls[b * H + k] + a1 * emb[b * H + k];
}

extern "C" void kernel_launch(void* const* d_in, const int* in_sizes, int n_in,
                              void* d_out, int out_size, void* d_ws, size_t ws_size,
                              hipStream_t stream) {
    const float* tokens = (const float*)d_in[0];
    const float* cls    = (const float*)d_in[1];
    const int*   mask   = (const int*)d_in[2];
    const float* W      = (const float*)d_in[3];
    const float* bias   = (const float*)d_in[4];
    const float* v      = (const float*)d_in[5];
    float* out = (float*)d_out;

    char* ws = (char*)d_ws;
    size_t off = 0;
    auto walloc = [&](size_t bytes) -> void* {
        void* p = ws + off;
        // +256 B pad per region: covers the final-iteration tile overfetch
        off = (off + bytes + 511) & ~(size_t)255;
        return p;
    };
    unsigned short* Wt    = (unsigned short*)walloc((size_t)H * K2 * 2);      // [H][K2]
    unsigned short* clsbf = (unsigned short*)walloc((size_t)BS * H * 2);
    unsigned short* Sbf   = (unsigned short*)walloc((size_t)2 * BS * H * 2);
    float* emb = (float*)walloc((size_t)BS * H * 4);
    float* P   = (float*)walloc((size_t)BS * H * 4);
    float* Q   = (float*)walloc((size_t)2 * BS * H * 4);

    transpose_convert_w<<<dim3(H / 32, K2 / 32), dim3(32, 8), 0, stream>>>(W, Wt);
    prep_kernel<<<BS, 256, 0, stream>>>(tokens, cls, mask, emb, clsbf, Sbf);
    // P = cls @ W_top : Bt rows are W columns, k in [0,768)
    gemm_bf16_kernel<<<dim3(H / 128, BS / 128), 256, 0, stream>>>(
        clsbf, H, Wt, K2, P, H);
    // Q = [cls;embraced] @ W_bot : k in [768,1536) -> Wt + H column offset
    gemm_bf16_kernel<<<dim3(H / 128, (2 * BS) / 128), 256, 0, stream>>>(
        Sbf, H, Wt + H, K2, Q, H);
    finalize_kernel<<<BS, 256, 0, stream>>>(P, Q, bias, v, cls, emb, out);
}
